// MultiHeadAttention_48971217109394
// MI455X (gfx1250) — matrix operand order repaired and checked by reading, not compiled, not run on hardware
//
#include <hip/hip_runtime.h>
#include <hip/hip_bf16.h>

// ---------------------------------------------------------------------------
// MI455X (gfx1250) Multi-Head Attention forward.
// All GEMMs use v_wmma_f32_16x16x32_bf16 (wave32, 16x16 tiles, f32 accum).
// alpha [B,H,S,S] must be materialized (it is a returned output), so the
// pipeline is: convert -> Q/K/V proj (WMMA) -> Vt transpose ->
// scores+mask (WMMA, causal tile skip) -> rowwise softmax (f32, LDS reduce)
// -> context = alpha @ V (WMMA, causal K-loop truncation) -> out proj (WMMA).
// ---------------------------------------------------------------------------

#define DEV __device__ __forceinline__

typedef __attribute__((ext_vector_type(16))) __bf16 v16bf;
typedef __attribute__((ext_vector_type(8)))  __bf16 v8bf;
typedef __attribute__((ext_vector_type(8)))  float  v8f;

static constexpr int   kB  = 2;
static constexpr int   kS  = 2048;
static constexpr int   kD  = 1024;
static constexpr int   kH  = 16;
static constexpr int   kDK = 64;
static constexpr float kNegInf = -1e30f;

DEV v8f wmma_bf16(v16bf a, v16bf b, v8f c) {
  // D(f32 16x16) = A(16x32 bf16) * B(32x16 bf16) + C
  return __builtin_amdgcn_wmma_f32_16x16x32_bf16(
      /*neg_a=*/false, a, /*neg_b=*/false, b,
      /*c_mod=*/(short)0, c, /*reuse_a=*/false, /*reuse_b=*/false);
}

// A fragment (16x32, bf16, row-major source, leading dim lda in elements).
// gfx1250 layout: lane<16 holds row m=lane, K = {k0..k0+7, k0+16..k0+23};
//                 lane>=16 holds row m=lane-16, K = {k0+8..+15, k0+24..+31}.
DEV v16bf load_a_frag(const __bf16* __restrict__ A, int lda, int row0, int k0,
                      int lane) {
  const int m  = row0 + (lane & 15);
  const int hi = lane >> 4;
  const __bf16* p = A + (size_t)m * lda + k0 + hi * 8;
  union { v16bf v; v8bf h[2]; } u;
  u.h[0] = *reinterpret_cast<const v8bf*>(p);        // 16B
  u.h[1] = *reinterpret_cast<const v8bf*>(p + 16);   // 16B
  return u.v;
}

// B fragment (32x16, bf16). Source is "output-column-major": row n of Wm is
// output column n (true for torch-Linear W, for K (keys), and for Vt).
// gfx1250 layout: lane<16 holds col n=n0+lane, K = {k0..k0+7, k0+16..k0+23};
//                 lane>=16 holds col n, K = {k0+8..+15, k0+24..+31}.
DEV v16bf load_b_frag(const __bf16* __restrict__ Wm, int ldw, int n0, int k0,
                      int lane) {
  const int n  = n0 + (lane & 15);
  const int hi = lane >> 4;
  const __bf16* p = Wm + (size_t)n * ldw + k0 + hi * 8;
  union { v16bf v; v8bf h[2]; } u;
  u.h[0] = *reinterpret_cast<const v8bf*>(p);        // 16B
  u.h[1] = *reinterpret_cast<const v8bf*>(p + 16);   // 16B
  return u.v;
}

// A fragment built from an f32 source (alpha), converted to bf16 in registers.
DEV v16bf load_a_frag_f32(const float* __restrict__ A, int lda, int row0,
                          int k0, int lane) {
  const int m  = row0 + (lane & 15);
  const int hi = lane >> 4;
  const float* p = A + (size_t)m * lda + k0 + hi * 8;
  const float4 c0 = reinterpret_cast<const float4*>(p)[0];
  const float4 c1 = reinterpret_cast<const float4*>(p)[1];
  const float4 c2 = reinterpret_cast<const float4*>(p + 16)[0];
  const float4 c3 = reinterpret_cast<const float4*>(p + 16)[1];
  v16bf v;
  v[0]  = (__bf16)c0.x; v[1]  = (__bf16)c0.y; v[2]  = (__bf16)c0.z; v[3]  = (__bf16)c0.w;
  v[4]  = (__bf16)c1.x; v[5]  = (__bf16)c1.y; v[6]  = (__bf16)c1.z; v[7]  = (__bf16)c1.w;
  v[8]  = (__bf16)c2.x; v[9]  = (__bf16)c2.y; v[10] = (__bf16)c2.z; v[11] = (__bf16)c2.w;
  v[12] = (__bf16)c3.x; v[13] = (__bf16)c3.y; v[14] = (__bf16)c3.z; v[15] = (__bf16)c3.w;
  return v;
}

// ---------------------------------------------------------------------------
// f32 -> bf16 conversion (X and the four weight matrices; 2 MB weights and
// 8 MB X stay resident in the 192 MB L2 for the GEMMs that follow).
// ---------------------------------------------------------------------------
__global__ void mha_cvt_bf16(const float* __restrict__ src,
                             __bf16* __restrict__ dst, int n) {
  int i = blockIdx.x * blockDim.x + threadIdx.x;
  if (i < n) dst[i] = (__bf16)src[i];
}

// V [B,S,D] bf16 -> Vt [B,H,DK,S] bf16 so columns of the context-GEMM B
// operand are contiguous rows.
__global__ void mha_transpose_v(const __bf16* __restrict__ Vb,
                                __bf16* __restrict__ Vt) {
  size_t i = blockIdx.x * (size_t)blockDim.x + threadIdx.x;
  const size_t total = (size_t)kB * kS * kD;
  if (i >= total) return;
  const int d = (int)(i & (kD - 1));
  const size_t bs = i >> 10;            // /kD
  const int s = (int)(bs & (kS - 1));
  const int b = (int)(bs >> 11);        // /kS
  const int h = d >> 6, dk = d & 63;
  Vt[(((size_t)(b * kH + h)) * kDK + dk) * kS + s] = Vb[i];
}

// ---------------------------------------------------------------------------
// C[M,N] = A[M,K](bf16) @ W[N,K]^T (bf16) + bias, output f32 or bf16.
// 128 threads = 4 waves; wave -> 32x32 subtile (2x2 WMMA); block -> 64x64.
// ---------------------------------------------------------------------------
template <bool OUT_F32>
__global__ __launch_bounds__(128) void mha_gemm_xwT(
    const __bf16* __restrict__ A, const __bf16* __restrict__ W,
    const float* __restrict__ bias, void* __restrict__ Cout,
    int K, int lda, int ldw, int ldc) {
  const int lane = threadIdx.x & 31;
  const int wave = threadIdx.x >> 5;
  const int row0 = blockIdx.y * 64 + (wave >> 1) * 32;
  const int col0 = blockIdx.x * 64 + (wave & 1) * 32;

  v8f acc[2][2] = {};
  for (int k0 = 0; k0 < K; k0 += 32) {
    if (k0 + 32 < K) {  // prefetch next K-tile of A (global_prefetch)
      __builtin_prefetch(
          (const void*)(A + (size_t)(row0 + (lane & 15)) * lda + k0 + 32), 0, 1);
    }
    const v16bf a0 = load_a_frag(A, lda, row0,      k0, lane);
    const v16bf a1 = load_a_frag(A, lda, row0 + 16, k0, lane);
    const v16bf b0 = load_b_frag(W, ldw, col0,      k0, lane);
    const v16bf b1 = load_b_frag(W, ldw, col0 + 16, k0, lane);
    acc[0][0] = wmma_bf16(a0, b0, acc[0][0]);
    acc[0][1] = wmma_bf16(a0, b1, acc[0][1]);
    acc[1][0] = wmma_bf16(a1, b0, acc[1][0]);
    acc[1][1] = wmma_bf16(a1, b1, acc[1][1]);
  }

  const int hi = lane >> 4;
#pragma unroll
  for (int i = 0; i < 2; ++i)
#pragma unroll
    for (int j = 0; j < 2; ++j) {
      const int col = col0 + j * 16 + (lane & 15);
      const float bv = bias[col];
      const int rbase = row0 + i * 16 + hi * 8;
#pragma unroll
      for (int r = 0; r < 8; ++r) {
        const float v = acc[i][j][r] + bv;
        const size_t idx = (size_t)(rbase + r) * ldc + col;
        if (OUT_F32) reinterpret_cast<float*>(Cout)[idx] = v;
        else         reinterpret_cast<__bf16*>(Cout)[idx] = (__bf16)v;
      }
    }
}

// ---------------------------------------------------------------------------
// scores[b,h,q,k] = (Q_h @ K_h^T) * 1/sqrt(dk), masked (causal + padding),
// written as f32 directly into the alpha output region (pre-softmax).
// Tiles entirely above the diagonal skip the GEMM and emit -1e30.
// ---------------------------------------------------------------------------
__global__ __launch_bounds__(128) void mha_scores(
    const __bf16* __restrict__ Qb, const __bf16* __restrict__ Kb,
    const int* __restrict__ pad, const int* __restrict__ causal_p,
    float* __restrict__ alpha) {
  const int lane = threadIdx.x & 31;
  const int wave = threadIdx.x >> 5;
  const int bh = blockIdx.z;                  // b*H + h
  const int b = bh >> 4, h = bh & 15;
  const int tq0 = blockIdx.y * 64, tc0 = blockIdx.x * 64;
  const int q0 = tq0 + (wave >> 1) * 32;
  const int c0 = tc0 + (wave & 1) * 32;
  const int causal = *causal_p;

  const __bf16* Qh = Qb + (size_t)b * kS * kD + h * kDK;  // row q: stride kD
  const __bf16* Kh = Kb + (size_t)b * kS * kD + h * kDK;
  float* out = alpha + (size_t)bh * kS * kS;
  const int* padb = pad + b * kS;

  const bool dead = (causal != 0) && (tc0 > tq0 + 63);  // uniform per block
  v8f acc[2][2] = {};
  if (!dead) {
#pragma unroll
    for (int k0 = 0; k0 < kDK; k0 += 32) {
      const v16bf a0 = load_a_frag(Qh, kD, q0,      k0, lane);
      const v16bf a1 = load_a_frag(Qh, kD, q0 + 16, k0, lane);
      const v16bf b0 = load_b_frag(Kh, kD, c0,      k0, lane);
      const v16bf b1 = load_b_frag(Kh, kD, c0 + 16, k0, lane);
      acc[0][0] = wmma_bf16(a0, b0, acc[0][0]);
      acc[0][1] = wmma_bf16(a0, b1, acc[0][1]);
      acc[1][0] = wmma_bf16(a1, b0, acc[1][0]);
      acc[1][1] = wmma_bf16(a1, b1, acc[1][1]);
    }
  }

  const float scale = 0.125f;  // 1/sqrt(64)
  const int hi = lane >> 4;
#pragma unroll
  for (int i = 0; i < 2; ++i)
#pragma unroll
    for (int j = 0; j < 2; ++j) {
      const int col = c0 + j * 16 + (lane & 15);
      const bool colpad = (padb[col] == 1);
      const int rbase = q0 + i * 16 + hi * 8;
#pragma unroll
      for (int r = 0; r < 8; ++r) {
        const int row = rbase + r;
        float v = dead ? kNegInf : acc[i][j][r] * scale;
        if ((causal && (col > row)) || colpad) v = kNegInf;
        out[(size_t)row * kS + col] = v;
      }
    }
}

// ---------------------------------------------------------------------------
// In-place row softmax over alpha: one 256-thread block per row (2048 elems,
// 8 per thread via 2x float4). f32 throughout; -1e30 entries -> exactly 0.
// ---------------------------------------------------------------------------
__global__ __launch_bounds__(256) void mha_softmax(float* __restrict__ alpha) {
  float* p = alpha + (size_t)blockIdx.x * kS;
  const int t = threadIdx.x;
  float4 v0 = reinterpret_cast<const float4*>(p)[2 * t];
  float4 v1 = reinterpret_cast<const float4*>(p)[2 * t + 1];
  float vals[8] = {v0.x, v0.y, v0.z, v0.w, v1.x, v1.y, v1.z, v1.w};

  __shared__ float red[256];
  float m = vals[0];
#pragma unroll
  for (int j = 1; j < 8; ++j) m = fmaxf(m, vals[j]);
  red[t] = m;
  __syncthreads();
  for (int off = 128; off > 0; off >>= 1) {
    if (t < off) red[t] = fmaxf(red[t], red[t + off]);
    __syncthreads();
  }
  const float mx = red[0];
  __syncthreads();

  float e[8], s = 0.f;
#pragma unroll
  for (int j = 0; j < 8; ++j) { e[j] = __expf(vals[j] - mx); s += e[j]; }
  red[t] = s;
  __syncthreads();
  for (int off = 128; off > 0; off >>= 1) {
    if (t < off) red[t] += red[t + off];
    __syncthreads();
  }
  const float inv = 1.0f / red[0];

  v0 = make_float4(e[0] * inv, e[1] * inv, e[2] * inv, e[3] * inv);
  v1 = make_float4(e[4] * inv, e[5] * inv, e[6] * inv, e[7] * inv);
  reinterpret_cast<float4*>(p)[2 * t]     = v0;
  reinterpret_cast<float4*>(p)[2 * t + 1] = v1;
}

// ---------------------------------------------------------------------------
// context[b,q,h*64+n] = sum_k alpha[b,h,q,k] * V[b,h,k,n]  (WMMA).
// alpha read as f32, converted to bf16 fragments in registers.
// Causal: alpha==0 for k>q, so truncate the K loop at the tile's max row.
// ---------------------------------------------------------------------------
__global__ __launch_bounds__(128) void mha_context(
    const float* __restrict__ alpha, const __bf16* __restrict__ Vt,
    const int* __restrict__ causal_p, __bf16* __restrict__ ctx) {
  const int lane = threadIdx.x & 31;
  const int wave = threadIdx.x >> 5;
  const int bh = blockIdx.y;
  const int b = bh >> 4, h = bh & 15;
  const int q0 = blockIdx.x * 64 + (wave >> 1) * 32;
  const int n0 = (wave & 1) * 32;  // 64 == kDK columns total

  const float*  Ar = alpha + (size_t)bh * kS * kS;   // [S,S], ld kS
  const __bf16* Bv = Vt + (size_t)bh * kDK * kS;     // row n = dk, ld kS
  const int causal = *causal_p;
  const int kmaxTile = blockIdx.x * 64 + 64;
  const int kmax = causal ? (kmaxTile < kS ? kmaxTile : kS) : kS;

  v8f acc[2][2] = {};
  for (int k0 = 0; k0 < kmax; k0 += 32) {
    const v16bf a0 = load_a_frag_f32(Ar, kS, q0,      k0, lane);
    const v16bf a1 = load_a_frag_f32(Ar, kS, q0 + 16, k0, lane);
    const v16bf b0 = load_b_frag(Bv, kS, n0,      k0, lane);
    const v16bf b1 = load_b_frag(Bv, kS, n0 + 16, k0, lane);
    acc[0][0] = wmma_bf16(a0, b0, acc[0][0]);
    acc[0][1] = wmma_bf16(a0, b1, acc[0][1]);
    acc[1][0] = wmma_bf16(a1, b0, acc[1][0]);
    acc[1][1] = wmma_bf16(a1, b1, acc[1][1]);
  }

  const int hi = lane >> 4;
#pragma unroll
  for (int i = 0; i < 2; ++i)
#pragma unroll
    for (int j = 0; j < 2; ++j) {
      const int n = n0 + j * 16 + (lane & 15);
      const int rbase = q0 + i * 16 + hi * 8;
#pragma unroll
      for (int r = 0; r < 8; ++r) {
        const int q = rbase + r;
        ctx[((size_t)b * kS + q) * kD + h * kDK + n] = (__bf16)acc[i][j][r];
      }
    }
}

// ---------------------------------------------------------------------------
extern "C" void kernel_launch(void* const* d_in, const int* in_sizes, int n_in,
                              void* d_out, int out_size, void* d_ws,
                              size_t ws_size, hipStream_t stream) {
  (void)in_sizes; (void)n_in; (void)out_size; (void)ws_size;

  const float* X      = (const float*)d_in[0];
  const int*   pad    = (const int*)d_in[1];
  const int*   causal = (const int*)d_in[2];
  const float* Wq = (const float*)d_in[3];  const float* bq = (const float*)d_in[4];
  const float* Wk = (const float*)d_in[5];  const float* bk = (const float*)d_in[6];
  const float* Wv = (const float*)d_in[7];  const float* bv = (const float*)d_in[8];
  const float* Wo = (const float*)d_in[9];  const float* bo = (const float*)d_in[10];

  float* out   = (float*)d_out;                        // [B,S,D]
  float* alpha = out + (size_t)kB * kS * kD;           // [B,H,S,S]

  const size_t MB = 1u << 20;
  char* ws = (char*)d_ws;
  __bf16* Xb  = (__bf16*)(ws);              //  8 MB  [B*S, D]
  __bf16* Wqb = (__bf16*)(ws +  8 * MB);    //  2 MB
  __bf16* Wkb = (__bf16*)(ws + 10 * MB);
  __bf16* Wvb = (__bf16*)(ws + 12 * MB);
  __bf16* Wob = (__bf16*)(ws + 14 * MB);
  __bf16* Qb  = (__bf16*)(ws + 16 * MB);    //  8 MB  [B*S, D]
  __bf16* Kb  = (__bf16*)(ws + 24 * MB);
  __bf16* Vb  = (__bf16*)(ws + 32 * MB);
  __bf16* Vt  = (__bf16*)(ws + 40 * MB);    //  8 MB  [B,H,DK,S]
  __bf16* Ctx = (__bf16*)(ws + 48 * MB);    //  8 MB  [B,S,D]

  const int nX = kB * kS * kD;      // 4,194,304
  const int nW = kD * kD;           // 1,048,576

  mha_cvt_bf16<<<(nX + 255) / 256, 256, 0, stream>>>(X,  Xb,  nX);
  mha_cvt_bf16<<<(nW + 255) / 256, 256, 0, stream>>>(Wq, Wqb, nW);
  mha_cvt_bf16<<<(nW + 255) / 256, 256, 0, stream>>>(Wk, Wkb, nW);
  mha_cvt_bf16<<<(nW + 255) / 256, 256, 0, stream>>>(Wv, Wvb, nW);
  mha_cvt_bf16<<<(nW + 255) / 256, 256, 0, stream>>>(Wo, Wob, nW);

  // Q/K/V projections: [4096,1024] = Xb @ W^T + b
  const dim3 gemmGrid(kD / 64, (kB * kS) / 64);   // (16, 64)
  mha_gemm_xwT<false><<<gemmGrid, 128, 0, stream>>>(Xb, Wqb, bq, Qb, kD, kD, kD, kD);
  mha_gemm_xwT<false><<<gemmGrid, 128, 0, stream>>>(Xb, Wkb, bk, Kb, kD, kD, kD, kD);
  mha_gemm_xwT<false><<<gemmGrid, 128, 0, stream>>>(Xb, Wvb, bv, Vb, kD, kD, kD, kD);

  mha_transpose_v<<<(nX + 255) / 256, 256, 0, stream>>>(Vb, Vt);

  // Scores + mask -> alpha (pre-softmax)
  const dim3 scoresGrid(kS / 64, kS / 64, kB * kH);  // (32, 32, 32)
  mha_scores<<<scoresGrid, 128, 0, stream>>>(Qb, Kb, pad, causal, alpha);

  // Row softmax in place: B*H*S rows
  mha_softmax<<<kB * kH * kS, 256, 0, stream>>>(alpha);

  // context = alpha @ V
  const dim3 ctxGrid(kS / 64, kB * kH);              // (32, 32)
  mha_context<<<ctxGrid, 128, 0, stream>>>(alpha, Vt, causal, Ctx);

  // output = context @ Wo^T + bo (f32 into d_out)
  mha_gemm_xwT<true><<<gemmGrid, 128, 0, stream>>>(Ctx, Wob, bo, out, kD, kD, kD, kD);
}
